// SelfAttentionBlock_34445637713961
// MI455X (gfx1250) — matrix operand order repaired and checked by reading, not compiled, not run on hardware
//
#include <hip/hip_runtime.h>
#include <cstdint>
#include <cstddef>

// ---------------------------------------------------------------------------
// Types for CDNA5 WMMA (gfx1250, wave32)
// ---------------------------------------------------------------------------
typedef __bf16 bf16_t;
typedef __attribute__((ext_vector_type(16))) __bf16 v16bf;
typedef __attribute__((ext_vector_type(8)))  float  v8f;

struct __align__(16) U4 { uint32_t x, y, z, w; };

__device__ __forceinline__ U4 ld4g(const bf16_t* p) {
  return *reinterpret_cast<const U4*>(p);
}
__device__ __forceinline__ U4 ld4s(const bf16_t* p) {
  return *reinterpret_cast<const U4*>(p);
}

__device__ __forceinline__ v8f wmma_bf16(v16bf a, v16bf b, v8f c) {
  // emits v_wmma_f32_16x16x32_bf16
  return __builtin_amdgcn_wmma_f32_16x16x32_bf16(false, a, false, b,
                                                 (short)0, c, false, false);
}

// ---------------------------------------------------------------------------
// Problem constants
// ---------------------------------------------------------------------------
#define NB   4
#define NC   256
#define NH   4
#define HD   64
#define NS   4096      // 64*64 tokens
#define NG   32
#define CPG  8         // channels per group
#define EPSV 1e-5f

// ---------------------------------------------------------------------------
// Workspace layout (bytes)
// ---------------------------------------------------------------------------
#define XNT_OFF   ((size_t)0)                     // bf16 [B][S][C]      8 MB
#define Q_OFF     ((size_t)8388608)               // bf16 [B][H][S][HD]  8 MB
#define K_OFF     ((size_t)16777216)              // bf16 [B][H][S][HD]  8 MB
#define VT_OFF    ((size_t)25165824)              // bf16 [B][H][HD][S]  8 MB
#define ATT_OFF   ((size_t)33554432)              // bf16 [B][S][C]      8 MB
#define WQ_OFF    ((size_t)41943040)              // bf16 [3C][C]
#define WP_OFF    ((size_t)42336256)              // bf16 [C][C]
#define ST_OFF    ((size_t)42467328)              // f32  [B*G][2]

// ---------------------------------------------------------------------------
// Kernel 1: GroupNorm statistics (one block per (b, g))
// ---------------------------------------------------------------------------
__global__ __launch_bounds__(256)
void gn_stats_kernel(const float* __restrict__ x, float* __restrict__ stats) {
  const int bg = blockIdx.x;                 // 0 .. NB*NG-1
  const int b = bg >> 5, g = bg & 31;
  const float* p = x + ((size_t)b * NC + (size_t)g * CPG) * NS;  // CPG*NS contiguous
  float s = 0.f, s2 = 0.f;
  for (int i = threadIdx.x; i < CPG * NS; i += 256) {
    float v = p[i];
    s += v; s2 += v * v;
  }
  __shared__ float sh0[256], sh1[256];
  sh0[threadIdx.x] = s; sh1[threadIdx.x] = s2;
  __syncthreads();
  for (int off = 128; off > 0; off >>= 1) {
    if (threadIdx.x < off) {
      sh0[threadIdx.x] += sh0[threadIdx.x + off];
      sh1[threadIdx.x] += sh1[threadIdx.x + off];
    }
    __syncthreads();
  }
  if (threadIdx.x == 0) {
    const float inv_n = 1.0f / (float)(CPG * NS);
    float mean = sh0[0] * inv_n;
    float var  = sh1[0] * inv_n - mean * mean;
    stats[bg * 2 + 0] = mean;
    stats[bg * 2 + 1] = rsqrtf(var + EPSV);
  }
}

// ---------------------------------------------------------------------------
// Kernel 2: normalize + gamma/beta, write token-major bf16 xnT[b][s][c]
// ---------------------------------------------------------------------------
__global__ __launch_bounds__(256)
void gn_apply_kernel(const float* __restrict__ x, const float* __restrict__ gamma,
                     const float* __restrict__ beta, const float* __restrict__ stats,
                     bf16_t* __restrict__ xnT) {
  const size_t i = (size_t)blockIdx.x * 256 + threadIdx.x;   // over B*C*S
  const int    sIdx = (int)(i & (NS - 1));
  const size_t bc   = i >> 12;
  const int    cIdx = (int)(bc & (NC - 1));
  const int    b    = (int)(bc >> 8);
  const int    g    = cIdx >> 3;
  const float mean = stats[(b * NG + g) * 2 + 0];
  const float rstd = stats[(b * NG + g) * 2 + 1];
  const float v = (x[i] - mean) * rstd * gamma[cIdx] + beta[cIdx];
  xnT[((size_t)b * NS + sIdx) * NC + cIdx] = (bf16_t)v;
}

// ---------------------------------------------------------------------------
// Kernel 3: convert weights to bf16
// ---------------------------------------------------------------------------
__global__ __launch_bounds__(256)
void cvt_w_kernel(const float* __restrict__ wqkv, const float* __restrict__ wproj,
                  bf16_t* __restrict__ wq_bf, bf16_t* __restrict__ wp_bf) {
  const int i = blockIdx.x * 256 + threadIdx.x;   // up to 3*C*C
  if (i < 3 * NC * NC) wq_bf[i] = (bf16_t)wqkv[i];
  if (i < NC * NC)     wp_bf[i] = (bf16_t)wproj[i];
}

// ---------------------------------------------------------------------------
// Kernel 4: QKV GEMM.  qkv[b,o,s] = sum_c wqkv[o,c] * xnT[b,s,c]
// One wave per 16x16 (o,s) tile; K=256 contraction in 8 chunks of 32.
// Q stored [b,h,s,d] (scaled by hd^-0.5), K stored [b,h,s,d], V stored [b,h,d,s].
// ---------------------------------------------------------------------------
__global__ __launch_bounds__(256)
void qkv_mm_kernel(const bf16_t* __restrict__ wq_bf, const bf16_t* __restrict__ xnT,
                   bf16_t* __restrict__ qb, bf16_t* __restrict__ kb,
                   bf16_t* __restrict__ vtb) {
  const int lane = threadIdx.x & 31;
  const int wave = threadIdx.x >> 5;
  const int tile = blockIdx.x * 8 + wave;     // 4 * 48 * 256 tiles
  const int st   = tile & 255;
  const int rest = tile >> 8;
  const int ot   = rest % 48;
  const int b    = rest / 48;

  const int col = lane & 15;
  const int rb8 = (lane < 16) ? 0 : 8;
  const int ka  = (lane < 16) ? 0 : 8;    // A-matrix K base
  const int kbb = (lane < 16) ? 0 : 16;   // unused (B loads use ka)

  const bf16_t* arow = wq_bf + (size_t)(ot * 16 + col) * NC;
  const bf16_t* brow = xnT + ((size_t)b * NS + st * 16 + col) * NC;

  v8f acc = {};
#pragma unroll
  for (int kc = 0; kc < NC; kc += 32) {
    v16bf a, bt;
    *((U4*)&a + 0) = ld4g(arow + kc + ka);
    *((U4*)&a + 1) = ld4g(arow + kc + ka + 16);
    *((U4*)&bt + 0) = ld4g(brow + kc + ka);
    *((U4*)&bt + 1) = ld4g(brow + kc + ka + 16);
    acc = wmma_bf16(a, bt, acc);
  }

#pragma unroll
  for (int r = 0; r < 8; ++r) {
    const int o = ot * 16 + r + rb8;
    const int s = st * 16 + col;
    float f = acc[r];
    if (o < NC) {                       // Q (fold in hd^-0.5 = 0.125)
      const int h = o >> 6, d = o & 63;
      qb[(((size_t)(b * NH + h) * NS) + s) * HD + d] = (bf16_t)(f * 0.125f);
    } else if (o < 2 * NC) {            // K
      const int oo = o - NC, h = oo >> 6, d = oo & 63;
      kb[(((size_t)(b * NH + h) * NS) + s) * HD + d] = (bf16_t)f;
    } else {                            // V, transposed [b,h,d,s]
      const int oo = o - 2 * NC, h = oo >> 6, d = oo & 63;
      vtb[(((size_t)(b * NH + h) * HD) + d) * NS + s] = (bf16_t)f;
    }
  }
}

// ---------------------------------------------------------------------------
// Kernel 5: flash attention.  One wave = 16 query rows; stream keys 32 at a time.
// Online softmax in f32; P re-laid out C->A via per-wave private LDS bounce.
// ---------------------------------------------------------------------------
__global__ __launch_bounds__(256)
void attn_kernel(const bf16_t* __restrict__ qb, const bf16_t* __restrict__ kb,
                 const bf16_t* __restrict__ vtb, bf16_t* __restrict__ attnT) {
  __shared__ bf16_t lds_p[8][16 * 32];        // per-wave private P staging (8 KB)

  const int lane = threadIdx.x & 31;
  const int wave = threadIdx.x >> 5;
  const int blk    = blockIdx.x;              // 0..511
  const int qchunk = blk & 31;                // S / 128
  const int bh     = blk >> 5;                // b*NH + h
  const int b = bh >> 2, h = bh & 3;
  const int q0 = qchunk * 128 + wave * 16;

  const int col = lane & 15;
  const int rb8 = (lane < 16) ? 0 : 8;
  const int ka  = (lane < 16) ? 0 : 8;
  const int kbb = (lane < 16) ? 0 : 16;

  const bf16_t* qptr = qb  + (size_t)bh * NS * HD;
  const bf16_t* kptr = kb  + (size_t)bh * NS * HD;
  const bf16_t* vptr = vtb + (size_t)bh * HD * NS;

  // Q A-tiles for d 0..31 and 32..63 (per lane: 8 d at ka, 8 at ka+16)
  v16bf qa0, qa1;
  {
    const bf16_t* qrow = qptr + (size_t)(q0 + col) * HD;
    *((U4*)&qa0 + 0) = ld4g(qrow + ka);
    *((U4*)&qa0 + 1) = ld4g(qrow + ka + 16);
    *((U4*)&qa1 + 0) = ld4g(qrow + 32 + ka);
    *((U4*)&qa1 + 1) = ld4g(qrow + 32 + ka + 16);
  }

  v8f acc[4] = {v8f{}, v8f{}, v8f{}, v8f{}};
  float m[8], l[8];
#pragma unroll
  for (int r = 0; r < 8; ++r) { m[r] = -1.0e30f; l[r] = 0.0f; }

  for (int kt = 0; kt < NS; kt += 32) {
    // prefetch next key/value tiles into cache (global_prefetch_b8)
    if (kt + 32 < NS) {
      __builtin_prefetch(kptr + (size_t)(kt + 32 + col) * HD, 0, 1);
      __builtin_prefetch(vptr + (size_t)col * NS + kt + 32, 0, 1);
    }

    // --- scores for keys kt..kt+31 (two 16-key C tiles) ---
    v8f stv[2];
#pragma unroll
    for (int hlf = 0; hlf < 2; ++hlf) {
      const bf16_t* krow = kptr + (size_t)(kt + hlf * 16 + col) * HD;
      v16bf kb0, kb1;
      *((U4*)&kb0 + 0) = ld4g(krow + ka);
      *((U4*)&kb0 + 1) = ld4g(krow + ka + 16);
      *((U4*)&kb1 + 0) = ld4g(krow + 32 + ka);
      *((U4*)&kb1 + 1) = ld4g(krow + 32 + ka + 16);
      v8f c = {};
      c = wmma_bf16(qa0, kb0, c);
      c = wmma_bf16(qa1, kb1, c);
      stv[hlf] = c;
    }

    // --- online softmax (row stats across 16 lanes via xor butterflies) ---
    float corr[8];
#pragma unroll
    for (int r = 0; r < 8; ++r) {
      float t = fmaxf(stv[0][r], stv[1][r]);
      t = fmaxf(t, __shfl_xor(t, 1, 32));
      t = fmaxf(t, __shfl_xor(t, 2, 32));
      t = fmaxf(t, __shfl_xor(t, 4, 32));
      t = fmaxf(t, __shfl_xor(t, 8, 32));
      const float mnew = fmaxf(m[r], t);
      corr[r] = __expf(m[r] - mnew);
      const float p0 = __expf(stv[0][r] - mnew);
      const float p1 = __expf(stv[1][r] - mnew);
      stv[0][r] = p0; stv[1][r] = p1;
      float s = p0 + p1;
      s += __shfl_xor(s, 1, 32);
      s += __shfl_xor(s, 2, 32);
      s += __shfl_xor(s, 4, 32);
      s += __shfl_xor(s, 8, 32);
      l[r] = l[r] * corr[r] + s;
      m[r] = mnew;
    }

    // --- rescale O accumulators, stage P tile in LDS (C layout -> memory) ---
#pragma unroll
    for (int r = 0; r < 8; ++r) {
      acc[0][r] *= corr[r];
      acc[1][r] *= corr[r];
      acc[2][r] *= corr[r];
      acc[3][r] *= corr[r];
      const int row = r + rb8;
      lds_p[wave][row * 32 + col]      = (bf16_t)stv[0][r];
      lds_p[wave][row * 32 + col + 16] = (bf16_t)stv[1][r];
    }
    asm volatile("s_wait_dscnt 0" ::: "memory");   // wave-synchronous LDS RAW

    // --- reload P as A-matrix 16x32 ---
    v16bf pa;
    {
      const bf16_t* pr = &lds_p[wave][(lane & 15) * 32];
      *((U4*)&pa + 0) = ld4s(pr + ka);
      *((U4*)&pa + 1) = ld4s(pr + ka + 16);
    }

    // --- O += P * V  (4 d-chunks of 16) ---
#pragma unroll
    for (int c = 0; c < 4; ++c) {
      const bf16_t* vrow = vptr + (size_t)(c * 16 + col) * NS + kt;
      v16bf vb;
      *((U4*)&vb + 0) = ld4g(vrow + ka);
      *((U4*)&vb + 1) = ld4g(vrow + ka + 16);
      acc[c] = wmma_bf16(pa, vb, acc[c]);
    }
  }

  // --- epilogue: normalize by l, store channel-last attnT[b][s][h*64+d] ---
  const size_t outBase = (size_t)b * NS * NC;
#pragma unroll
  for (int c = 0; c < 4; ++c) {
#pragma unroll
    for (int r = 0; r < 8; ++r) {
      const int srow = q0 + r + rb8;
      const int ch   = h * 64 + c * 16 + col;
      attnT[outBase + (size_t)srow * NC + ch] = (bf16_t)(acc[c][r] / l[r]);
    }
  }
}

// ---------------------------------------------------------------------------
// Kernel 6: output projection + bias + residual (f32 output)
// ---------------------------------------------------------------------------
__global__ __launch_bounds__(256)
void proj_mm_kernel(const bf16_t* __restrict__ wp_bf, const bf16_t* __restrict__ attnT,
                    const float* __restrict__ x, const float* __restrict__ bproj,
                    float* __restrict__ out) {
  const int lane = threadIdx.x & 31;
  const int wave = threadIdx.x >> 5;
  const int tile = blockIdx.x * 8 + wave;     // 4 * 16 * 256 tiles
  const int st   = tile & 255;
  const int rest = tile >> 8;
  const int ot   = rest & 15;
  const int b    = rest >> 4;

  const int col = lane & 15;
  const int rb8 = (lane < 16) ? 0 : 8;
  const int ka  = (lane < 16) ? 0 : 8;
  const int kbb = (lane < 16) ? 0 : 16;

  const bf16_t* arow = wp_bf + (size_t)(ot * 16 + col) * NC;
  const bf16_t* brow = attnT + ((size_t)b * NS + st * 16 + col) * NC;

  v8f acc = {};
#pragma unroll
  for (int kc = 0; kc < NC; kc += 32) {
    v16bf a, bt;
    *((U4*)&a + 0) = ld4g(arow + kc + ka);
    *((U4*)&a + 1) = ld4g(arow + kc + ka + 16);
    *((U4*)&bt + 0) = ld4g(brow + kc + ka);
    *((U4*)&bt + 1) = ld4g(brow + kc + ka + 16);
    acc = wmma_bf16(a, bt, acc);
  }

#pragma unroll
  for (int r = 0; r < 8; ++r) {
    const int o = ot * 16 + r + rb8;
    const int s = st * 16 + col;
    const size_t idx = ((size_t)(b * NC + o)) * NS + s;
    out[idx] = x[idx] + acc[r] + bproj[o];
  }
}

// ---------------------------------------------------------------------------
// Launcher
// ---------------------------------------------------------------------------
extern "C" void kernel_launch(void* const* d_in, const int* in_sizes, int n_in,
                              void* d_out, int out_size, void* d_ws, size_t ws_size,
                              hipStream_t stream) {
  const float* x      = (const float*)d_in[0];
  const float* gamma  = (const float*)d_in[1];
  const float* beta   = (const float*)d_in[2];
  const float* w_qkv  = (const float*)d_in[3];
  const float* w_proj = (const float*)d_in[4];
  const float* b_proj = (const float*)d_in[5];
  float* out = (float*)d_out;

  char* ws = (char*)d_ws;
  bf16_t* xnT   = (bf16_t*)(ws + XNT_OFF);
  bf16_t* qb    = (bf16_t*)(ws + Q_OFF);
  bf16_t* kbuf  = (bf16_t*)(ws + K_OFF);
  bf16_t* vtb   = (bf16_t*)(ws + VT_OFF);
  bf16_t* attnT = (bf16_t*)(ws + ATT_OFF);
  bf16_t* wq_bf = (bf16_t*)(ws + WQ_OFF);
  bf16_t* wp_bf = (bf16_t*)(ws + WP_OFF);
  float*  stats = (float*)(ws + ST_OFF);

  gn_stats_kernel<<<NB * NG, 256, 0, stream>>>(x, stats);
  gn_apply_kernel<<<(NB * NC * NS) / 256, 256, 0, stream>>>(x, gamma, beta, stats, xnT);
  cvt_w_kernel<<<(3 * NC * NC) / 256, 256, 0, stream>>>(w_qkv, w_proj, wq_bf, wp_bf);
  qkv_mm_kernel<<<(NB * 48 * 256) / 8, 256, 0, stream>>>(wq_bf, xnT, qb, kbuf, vtb);
  attn_kernel<<<NB * NH * (NS / 128), 256, 0, stream>>>(qb, kbuf, vtb, attnT);
  proj_mm_kernel<<<(NB * 16 * 256) / 8, 256, 0, stream>>>(wp_bf, attnT, x, b_proj, out);
}
